// MedicalAttention_17145509446366
// MI455X (gfx1250) — matrix operand order repaired and checked by reading, not compiled, not run on hardware
//
#include <hip/hip_runtime.h>

// ---------------- types ----------------
typedef __bf16 bf16;
typedef __attribute__((ext_vector_type(16))) __bf16 v16bf;
typedef __attribute__((ext_vector_type(2)))  __bf16 bf16x2;
typedef __attribute__((ext_vector_type(8)))  float  v8f;

union FragU { v16bf v; bf16x2 p[8]; };

__device__ __forceinline__ v8f v8f_zero() {
  v8f z;
#pragma unroll
  for (int i = 0; i < 8; ++i) z[i] = 0.0f;
  return z;
}

#define WMMA_BF16(a, b, c) \
  __builtin_amdgcn_wmma_f32_16x16x32_bf16(false, (a), false, (b), (short)0, (c), false, false)

// CDNA5 async global->LDS copy: 32 bytes (2 x B128) per call.
// VDST carries the LDS byte address; INST_OFFSET is added to BOTH lds and
// global addresses (CDNA5 ISA), so one offset covers both.
__device__ __forceinline__ void async_b128x2(unsigned lds_addr, const bf16* g) {
  asm volatile("global_load_async_to_lds_b128 %0, %1, off\n\t"
               "global_load_async_to_lds_b128 %0, %1, off offset:16"
               :: "v"(lds_addr), "v"(g) : "memory");
}
__device__ __forceinline__ void wait_async0() {
  asm volatile("s_wait_asynccnt 0x0" ::: "memory");
}
__device__ __forceinline__ unsigned lds_addr_of(const void* p) {
  // low 32 bits of a generic pointer into LDS == LDS byte address (flat aperture rule)
  return (unsigned)(uintptr_t)p;
}

// Problem constants
#define BB   4
#define SS   2048
#define HIDD 1024
#define NHH  16
#define HDD  64

// ---------------- f32 -> bf16 conversion (packed) ----------------
__global__ __launch_bounds__(256) void f32_to_bf16(const float* __restrict__ src,
                                                   bf16* __restrict__ dst, int npairs) {
  int i = blockIdx.x * blockDim.x + threadIdx.x;
  int stride = gridDim.x * blockDim.x;
  const float2* s2 = (const float2*)src;
  bf16x2* d2 = (bf16x2*)dst;
  for (; i < npairs; i += stride) {
    float2 f = s2[i];
    bf16x2 o; o[0] = (bf16)f.x; o[1] = (bf16)f.y;
    d2[i] = o;
  }
}

// f32 [K][N] -> bf16 transposed [N][K]  (weights; one-time, write-coalesced)
__global__ __launch_bounds__(256) void f32_to_bf16_T(const float* __restrict__ src,
                                                     bf16* __restrict__ dst) {
  int i = blockIdx.x * 256 + threadIdx.x;  // 0 .. HIDD*HIDD-1
  int n = i >> 10, kx = i & (HIDD - 1);
  dst[i] = (bf16)src[(size_t)kx * HIDD + n];
}

// ---------------- bf16 WMMA GEMM ----------------
// C[M=8192, N=1024] = A[M,1024] * Wt[N,1024]^T (+bias)
// mode 0: (acc+bias)*scale -> bf16 head layout [b,h,s,hd]        (Q, K)
// mode 2: (acc+bias)*scale -> bf16 head-transposed [b,h,hd,s]    (V)
// mode 1: acc+bias+residual -> f32 plain row-major               (out proj)
__global__ __launch_bounds__(256) void gemm_bf16(const bf16* __restrict__ A,
                                                 const bf16* __restrict__ Wt,
                                                 const float* __restrict__ bias,
                                                 bf16* __restrict__ dst_bf16,
                                                 float* __restrict__ dst_f32,
                                                 const float* __restrict__ residual,
                                                 float scale, int mode) {
  __shared__ __align__(16) bf16 ldsA[2][128 * 32];  // [m][k]
  __shared__ __align__(16) bf16 ldsB[2][128 * 32];  // [n][k]

  const int tid = threadIdx.x;
  const int m0 = blockIdx.y * 128;
  const int n0 = blockIdx.x * 128;
  const int w = tid >> 5;
  const int lane = tid & 31;
  const int g = lane >> 4;       // 16-lane half
  const int l = lane & 15;
  const int wm = (w & 3) * 32;   // 4 waves down
  const int wn = (w >> 2) * 64;  // 2 waves across

  // staging assignment: 128 rows x 32 cols per tile, 32B (16 bf16) per thread
  const int srow = tid >> 1;
  const int shalf = (tid & 1) * 16;
  const bf16* aRow = A + (size_t)(m0 + srow) * HIDD + shalf;
  const bf16* bRow = Wt + (size_t)(n0 + srow) * HIDD + shalf;

  v8f acc[2][4];
#pragma unroll
  for (int i = 0; i < 2; ++i)
#pragma unroll
    for (int j = 0; j < 4; ++j) acc[i][j] = v8f_zero();

  // prologue: stage k-step 0 into buffer 0
  async_b128x2(lds_addr_of(&ldsA[0][srow * 32 + shalf]), aRow);
  async_b128x2(lds_addr_of(&ldsB[0][srow * 32 + shalf]), bRow);
  wait_async0();
  __syncthreads();

  int buf = 0;
  for (int kk = 0; kk < HIDD; kk += 32) {
    // prefetch next k-step into the other buffer while we compute
    if (kk + 32 < HIDD) {
      async_b128x2(lds_addr_of(&ldsA[buf ^ 1][srow * 32 + shalf]), aRow + kk + 32);
      async_b128x2(lds_addr_of(&ldsB[buf ^ 1][srow * 32 + shalf]), bRow + kk + 32);
    }

    FragU af[2], bu[4];
#pragma unroll
    for (int mt = 0; mt < 2; ++mt) {
      int row = wm + mt * 16 + l;
#pragma unroll
      for (int vv = 0; vv < 8; ++vv) {
        int kb = (vv < 4) ? (g * 8 + 2 * vv) : (g * 8 + 2 * vv + 8);  // ISA A-frag map
        af[mt].p[vv] = *(const bf16x2*)&ldsA[buf][row * 32 + kb];
      }
    }
#pragma unroll
    for (int nt = 0; nt < 4; ++nt) {
      int col = wn + nt * 16 + l;
#pragma unroll
      for (int vv = 0; vv < 8; ++vv)  // B-frag: same K map as A-frag, packed pairs
        bu[nt].p[vv] = *(const bf16x2*)&ldsB[buf][col * 32 + g * 8 + 2 * vv + ((vv < 4) ? 0 : 8)];
    }
#pragma unroll
    for (int mt = 0; mt < 2; ++mt)
#pragma unroll
      for (int nt = 0; nt < 4; ++nt)
        acc[mt][nt] = WMMA_BF16(af[mt].v, bu[nt].v, acc[mt][nt]);

    wait_async0();     // next buffer resident (this wave's async ops)
    __syncthreads();   // all waves done reading current buffer + all writes visible
    buf ^= 1;
  }

  // epilogue; C layout: VGPR r -> row (r + 8*g), lane l -> col
#pragma unroll
  for (int mt = 0; mt < 2; ++mt) {
#pragma unroll
    for (int nt = 0; nt < 4; ++nt) {
      int ncol = n0 + wn + nt * 16 + l;
      float bnc = bias[ncol];
#pragma unroll
      for (int r = 0; r < 8; ++r) {
        int mrow = m0 + wm + mt * 16 + r + 8 * g;
        float vv = acc[mt][nt][r] + bnc;
        int bb = mrow >> 11, ss = mrow & (SS - 1);
        int hh = ncol >> 6, dd = ncol & (HDD - 1);
        if (mode == 0) {
          dst_bf16[(((size_t)(bb * NHH + hh) * SS) + ss) * HDD + dd] = (bf16)(vv * scale);
        } else if (mode == 2) {
          dst_bf16[(((size_t)(bb * NHH + hh) * HDD) + dd) * SS + ss] = (bf16)(vv * scale);
        } else {
          size_t idx = (size_t)mrow * HIDD + ncol;
          dst_f32[idx] = vv + residual[idx];
        }
      }
    }
  }
}

// ---------------- flash attention ----------------
// grid (S/64, NH, B), 128 threads (4 waves); wave w owns 16 query rows.
// q,k in [b,h,s,d]; v pre-transposed to [b,h,d,s] so all staging is row-major async.
__global__ __launch_bounds__(128) void flash_attn(const bf16* __restrict__ q,
                                                  const bf16* __restrict__ k,
                                                  const bf16* __restrict__ vT,
                                                  const float* __restrict__ attn_mask,
                                                  const float* __restrict__ med_mask,
                                                  const float* __restrict__ med_rel,
                                                  bf16* __restrict__ ctx) {
  __shared__ __align__(16) bf16 ldsK[2][32 * 64];    // [key][d]
  __shared__ __align__(16) bf16 ldsVt[2][64 * 32];   // [d][key]
  __shared__ __align__(16) bf16 ldsP[4][16 * 32];    // per-wave P tile [m][key]

  const int tid = threadIdx.x;
  const int w = tid >> 5;
  const int lane = tid & 31;
  const int g = lane >> 4;
  const int l = lane & 15;
  const int h = blockIdx.y, b = blockIdx.z;
  const int bh = b * NHH + h;
  const int q0 = blockIdx.x * 64 + w * 16;

  const bf16* kbase = k + (size_t)bh * SS * HDD;
  const bf16* vtbase = vT + (size_t)bh * HDD * SS;

  // staging assignment: both tiles are 2048 bf16 = 4KB; 32B per thread
  const int krow = tid >> 2, kseg = (tid & 3) * 16;  // K tile: 32 rows x 64
  const int vrow = tid >> 1, vseg = (tid & 1) * 16;  // Vt tile: 64 rows x 32

  // Q fragments (two K-halves of head_dim 64); Q already pre-scaled by 1/8
  FragU qa[2];
  const bf16* qrow = q + ((size_t)bh * SS + q0 + l) * HDD;
#pragma unroll
  for (int kh = 0; kh < 2; ++kh)
#pragma unroll
    for (int vv = 0; vv < 8; ++vv) {
      int kb = (vv < 4) ? (g * 8 + 2 * vv) : (g * 8 + 2 * vv + 8);
      qa[kh].p[vv] = *(const bf16x2*)(qrow + kh * 32 + kb);
    }

  float medq[8];
#pragma unroll
  for (int r = 0; r < 8; ++r) medq[r] = med_mask[(size_t)b * SS + q0 + r + 8 * g];

  float mi[8], li[8];
  v8f o[4];
#pragma unroll
  for (int r = 0; r < 8; ++r) { mi[r] = -1e30f; li[r] = 0.0f; }
#pragma unroll
  for (int nt = 0; nt < 4; ++nt) o[nt] = v8f_zero();

  // prologue: stage key-block 0
  async_b128x2(lds_addr_of(&ldsK[0][krow * 64 + kseg]), kbase + (size_t)krow * HDD + kseg);
  async_b128x2(lds_addr_of(&ldsVt[0][vrow * 32 + vseg]), vtbase + (size_t)vrow * SS + vseg);
  wait_async0();
  __syncthreads();

  int buf = 0;
  for (int jb = 0; jb < SS / 32; ++jb) {
    const int key0 = jb * 32;
    if (jb + 1 < SS / 32) {  // prefetch next key block
      const int kn = key0 + 32;
      async_b128x2(lds_addr_of(&ldsK[buf ^ 1][krow * 64 + kseg]),
                   kbase + (size_t)(kn + krow) * HDD + kseg);
      async_b128x2(lds_addr_of(&ldsVt[buf ^ 1][vrow * 32 + vseg]),
                   vtbase + (size_t)vrow * SS + kn + vseg);
    }

    // scores: S = Q * K^T  (two 16-key strips)
    v8f sacc[2];
    sacc[0] = v8f_zero();
    sacc[1] = v8f_zero();
#pragma unroll
    for (int nt = 0; nt < 2; ++nt) {
      int col = nt * 16 + l;  // key within block
#pragma unroll
      for (int kh = 0; kh < 2; ++kh) {
        FragU kf;
#pragma unroll
        for (int vv = 0; vv < 8; ++vv)
          kf.p[vv] = *(const bf16x2*)&ldsK[buf][col * 64 + kh * 32 + g * 8 + 2 * vv + ((vv < 4) ? 0 : 8)];
        sacc[nt] = WMMA_BF16(qa[kh].v, kf.v, sacc[nt]);
      }
    }

    // bias: attention mask + (sparse) medical boost (direct coalesced loads)
#pragma unroll
    for (int nt = 0; nt < 2; ++nt) {
      int keyg = key0 + nt * 16 + l;
      float ma = (1.0f - attn_mask[(size_t)b * SS + keyg]) * -10000.0f;
      float mk = med_mask[(size_t)b * SS + keyg];
#pragma unroll
      for (int r = 0; r < 8; ++r) {
        float sc = sacc[nt][r] + ma;
        float mm = medq[r] * mk;
        if (mm != 0.0f)  // exec-masked med_rel load (~4% of lanes)
          sc += mm * 0.1f * med_rel[(size_t)(q0 + r + 8 * g) * SS + keyg];
        sacc[nt][r] = sc;
      }
    }

    // online softmax; row r+8g lives across one 16-lane half
    float bm[8];
#pragma unroll
    for (int r = 0; r < 8; ++r) bm[r] = fmaxf(sacc[0][r], sacc[1][r]);
#pragma unroll
    for (int off = 1; off < 16; off <<= 1)
#pragma unroll
      for (int r = 0; r < 8; ++r) bm[r] = fmaxf(bm[r], __shfl_xor(bm[r], off, 32));

    float alpha[8], rs[8];
#pragma unroll
    for (int r = 0; r < 8; ++r) {
      float nm = fmaxf(mi[r], bm[r]);
      alpha[r] = __expf(mi[r] - nm);
      mi[r] = nm;
      rs[r] = 0.0f;
    }
#pragma unroll
    for (int nt = 0; nt < 2; ++nt)
#pragma unroll
      for (int r = 0; r < 8; ++r) {
        float p = __expf(sacc[nt][r] - mi[r]);
        rs[r] += p;
        ldsP[w][(r + 8 * g) * 32 + nt * 16 + l] = (bf16)p;  // C-layout -> [m][key]
      }
#pragma unroll
    for (int off = 1; off < 16; off <<= 1)
#pragma unroll
      for (int r = 0; r < 8; ++r) rs[r] += __shfl_xor(rs[r], off, 32);
#pragma unroll
    for (int r = 0; r < 8; ++r) li[r] = li[r] * alpha[r] + rs[r];
#pragma unroll
    for (int nt = 0; nt < 4; ++nt)
#pragma unroll
      for (int r = 0; r < 8; ++r) o[nt][r] *= alpha[r];

    // reload P as A-fragment (in-wave LDS round trip; DS ops are in-order)
    FragU pf;
#pragma unroll
    for (int vv = 0; vv < 8; ++vv) {
      int kb = (vv < 4) ? (g * 8 + 2 * vv) : (g * 8 + 2 * vv + 8);
      pf.p[vv] = *(const bf16x2*)&ldsP[w][l * 32 + kb];
    }
    // O += P * V  (4 dim-strips of 16)
#pragma unroll
    for (int nt = 0; nt < 4; ++nt) {
      FragU vf;
#pragma unroll
      for (int vv = 0; vv < 8; ++vv)
        vf.p[vv] = *(const bf16x2*)&ldsVt[buf][(nt * 16 + l) * 32 + g * 8 + 2 * vv + ((vv < 4) ? 0 : 8)];
      o[nt] = WMMA_BF16(pf.v, vf.v, o[nt]);
    }

    wait_async0();
    __syncthreads();
    buf ^= 1;
  }

  // epilogue: normalize, write ctx in [b, s, h, d] layout (== reshape to [b,s,hid])
#pragma unroll
  for (int r = 0; r < 8; ++r) {
    float inv = 1.0f / li[r];
    int srow = q0 + r + 8 * g;
#pragma unroll
    for (int nt = 0; nt < 4; ++nt)
      ctx[(((size_t)b * SS + srow) * NHH + h) * HDD + nt * 16 + l] = (bf16)(o[nt][r] * inv);
  }
}

// ---------------- layernorm (in-place over f32 rows) ----------------
__global__ __launch_bounds__(256) void layer_norm(float* __restrict__ x,
                                                  const float* __restrict__ gamma,
                                                  const float* __restrict__ beta) {
  __shared__ float red[8];
  const int tid = threadIdx.x;
  const int w = tid >> 5, lane = tid & 31;
  float* xr = x + (size_t)blockIdx.x * HIDD;

  float vals[4];
  float s = 0.0f;
#pragma unroll
  for (int i = 0; i < 4; ++i) { vals[i] = xr[tid + i * 256]; s += vals[i]; }
#pragma unroll
  for (int off = 1; off < 32; off <<= 1) s += __shfl_xor(s, off, 32);
  if (lane == 0) red[w] = s;
  __syncthreads();
  float tot = 0.0f;
#pragma unroll
  for (int i = 0; i < 8; ++i) tot += red[i];
  float mu = tot * (1.0f / HIDD);
  __syncthreads();

  float vs = 0.0f;
#pragma unroll
  for (int i = 0; i < 4; ++i) { float d = vals[i] - mu; vs += d * d; }
#pragma unroll
  for (int off = 1; off < 32; off <<= 1) vs += __shfl_xor(vs, off, 32);
  if (lane == 0) red[w] = vs;
  __syncthreads();
  float vtot = 0.0f;
#pragma unroll
  for (int i = 0; i < 8; ++i) vtot += red[i];
  float rstd = rsqrtf(vtot * (1.0f / HIDD) + 1e-12f);
#pragma unroll
  for (int i = 0; i < 4; ++i) {
    int c = tid + i * 256;
    xr[c] = (vals[i] - mu) * rstd * gamma[c] + beta[c];
  }
}

// ---------------- launcher ----------------
extern "C" void kernel_launch(void* const* d_in, const int* in_sizes, int n_in,
                              void* d_out, int out_size, void* d_ws, size_t ws_size,
                              hipStream_t stream) {
  (void)in_sizes; (void)n_in; (void)out_size; (void)ws_size;
  const float* hs      = (const float*)d_in[0];
  const float* amask   = (const float*)d_in[1];
  const float* mmask   = (const float*)d_in[2];
  const float* Wq      = (const float*)d_in[3];
  const float* bq      = (const float*)d_in[4];
  const float* Wk      = (const float*)d_in[5];
  const float* bk      = (const float*)d_in[6];
  const float* Wv      = (const float*)d_in[7];
  const float* bv      = (const float*)d_in[8];
  const float* med_rel = (const float*)d_in[9];
  const float* Wd      = (const float*)d_in[10];
  const float* bd      = (const float*)d_in[11];
  const float* ln_g    = (const float*)d_in[12];
  const float* ln_b    = (const float*)d_in[13];
  float* out = (float*)d_out;

  const size_t HSN = (size_t)BB * SS * HIDD;  // 8,388,608
  const size_t WN  = (size_t)HIDD * HIDD;     // 1,048,576
  bf16* hsb  = (bf16*)d_ws;
  bf16* wqb  = hsb + HSN;   // transposed bf16 weights [N][K]
  bf16* wkb  = wqb + WN;
  bf16* wvb  = wkb + WN;
  bf16* wdb  = wvb + WN;
  bf16* qb   = wdb + WN;    // [b,h,s,d]
  bf16* kb   = qb + HSN;    // [b,h,s,d]
  bf16* vtb  = kb + HSN;    // [b,h,d,s]
  bf16* ctxb = vtb + HSN;   // total ~88 MB of d_ws

  f32_to_bf16<<<2048, 256, 0, stream>>>(hs, hsb, (int)(HSN / 2));
  f32_to_bf16_T<<<4096, 256, 0, stream>>>(Wq, wqb);
  f32_to_bf16_T<<<4096, 256, 0, stream>>>(Wk, wkb);
  f32_to_bf16_T<<<4096, 256, 0, stream>>>(Wv, wvb);
  f32_to_bf16_T<<<4096, 256, 0, stream>>>(Wd, wdb);

  dim3 gg(HIDD / 128, (BB * SS) / 128);  // (8, 64)
  // Q gets the 1/sqrt(HD)=0.125 attention scale folded in
  gemm_bf16<<<gg, 256, 0, stream>>>(hsb, wqb, bq, qb, nullptr, nullptr, 0.125f, 0);
  gemm_bf16<<<gg, 256, 0, stream>>>(hsb, wkb, bk, kb, nullptr, nullptr, 1.0f, 0);
  gemm_bf16<<<gg, 256, 0, stream>>>(hsb, wvb, bv, vtb, nullptr, nullptr, 1.0f, 2);

  flash_attn<<<dim3(SS / 64, NHH, BB), 128, 0, stream>>>(qb, kb, vtb, amask, mmask, med_rel, ctxb);

  // out projection + bias + residual -> f32 in d_out
  gemm_bf16<<<gg, 256, 0, stream>>>(ctxb, wdb, bd, nullptr, out, hs, 1.0f, 1);

  layer_norm<<<BB * SS, 256, 0, stream>>>(out, ln_g, ln_b);
}
